// SimpleNLPModel_257698038099
// MI455X (gfx1250) — compile-verified, matrix operand order checked by reading, not run on hardware
//
#include <hip/hip_runtime.h>
#include <hip/hip_bf16.h>

typedef __attribute__((ext_vector_type(16))) _Float16 v16h;
typedef __attribute__((ext_vector_type(8)))  float    v8f;

// ---------------------------------------------------------------------------
// Kernel 1: embedding gather + mean pool.
// One block per batch row. 128 threads = 4 waves (wave32).
// Wave g handles tokens l = g, g+4, g+8, ... (50 tokens each).
// Lane holds a float4 slice of the 128-dim embedding (32 lanes * 4 = 128).
// Token-index loads are uniform within a wave -> scalar loads.
// ---------------------------------------------------------------------------
__global__ __launch_bounds__(128) void gather_pool_kernel(
    const int* __restrict__ x,          // [B, 200] token ids
    const float* __restrict__ emb,      // [VOCAB, 128]
    float* __restrict__ pooled)         // [B, 128]
{
    __shared__ float4 part[128];

    const int b    = blockIdx.x;
    const int t    = threadIdx.x;
    const int g    = t >> 5;            // wave id (token phase), 0..3
    const int lane = t & 31;

    const long long xbase = (long long)b * 200;
    float4 acc = make_float4(0.f, 0.f, 0.f, 0.f);

    #pragma unroll 5
    for (int l = g; l < 200; l += 4) {
        const int idx = x[xbase + l];   // wave-uniform -> s_load
        const float4 e = *(const float4*)(emb + (long long)idx * 128 + lane * 4);
        acc.x += e.x; acc.y += e.y; acc.z += e.z; acc.w += e.w;
    }

    part[t] = acc;
    __syncthreads();

    if (g == 0) {
        const float4 a0 = part[lane];
        const float4 a1 = part[32 + lane];
        const float4 a2 = part[64 + lane];
        const float4 a3 = part[96 + lane];
        const float inv = 1.0f / 200.0f;
        float4 r;
        r.x = (a0.x + a1.x + a2.x + a3.x) * inv;
        r.y = (a0.y + a1.y + a2.y + a3.y) * inv;
        r.z = (a0.z + a1.z + a2.z + a3.z) * inv;
        r.w = (a0.w + a1.w + a2.w + a3.w) * inv;
        *(float4*)(pooled + (long long)b * 128 + lane * 4) = r;
    }
}

// ---------------------------------------------------------------------------
// Kernel 2: h = relu(pooled @ W1 + b1); out = h @ W2 + b2
// One block per 16-row tile of pooled. 256 threads = 8 waves.
// Wave w computes the 16x16 tile of h at columns [16w, 16w+16) using
// v_wmma_f32_16x16x32_f16, K=128 in 4 chunks of 32.
//
// CDNA5 wave32 fragment layout (ISA 7.12.2):
//   A (16x32 f16): lane L -> row M = L%16; kb = (L<16 ? 0 : 8)
//     halves a[0..7]  = K = kb + 0..7
//     halves a[8..15] = K = kb + 16..23
//   B (32x16 f16): mirrored, lane L -> col N = L%16, same K pattern.
//   C/D (16x16 f32, 8 VGPRs): c[r] = D[M = r + 8*(L/16)][N = L%16]
// ---------------------------------------------------------------------------
__global__ __launch_bounds__(256) void mlp_wmma_kernel(
    const float* __restrict__ pooled,   // [B, 128]
    const float* __restrict__ W1,       // [128, 128]
    const float* __restrict__ b1,       // [128]
    const float* __restrict__ W2,       // [128, 2]
    const float* __restrict__ b2,       // [2]
    float* __restrict__ out)            // [B, 2]
{
    __shared__ float h_lds[16 * 128];

    const int r0   = blockIdx.x * 16;   // first batch row of this tile
    const int wave = threadIdx.x >> 5;  // 0..7 -> output column tile
    const int lane = threadIdx.x & 31;
    const int n0   = wave * 16;
    const int mn   = lane & 15;         // A row within tile / B col within tile
    const int hi   = lane >> 4;         // 0 or 1 -> K sub-offset / D row offset

    v8f acc = {};

    const float* arow = pooled + (long long)(r0 + mn) * 128;

    #pragma unroll
    for (int kc = 0; kc < 4; ++kc) {
        const int kb = kc * 32 + hi * 8;
        v16h a, bm;
        #pragma unroll
        for (int j = 0; j < 8; ++j) {
            a[j]      = (_Float16)arow[kb + j];
            a[j + 8]  = (_Float16)arow[kb + 16 + j];
            bm[j]     = (_Float16)W1[(kb + j) * 128 + (n0 + mn)];
            bm[j + 8] = (_Float16)W1[(kb + 16 + j) * 128 + (n0 + mn)];
        }
        // (neg_a, A, neg_b, B, c_mod, C, reuse_a, reuse_b)
        acc = __builtin_amdgcn_wmma_f32_16x16x32_f16(
            false, a, false, bm, (short)0, acc, false, false);
    }

    // bias + relu, stage h tile in LDS
    const float bias = b1[n0 + mn];
    #pragma unroll
    for (int r = 0; r < 8; ++r) {
        const int row = r + hi * 8;
        float v = acc[r] + bias;
        v = v > 0.0f ? v : 0.0f;
        h_lds[row * 128 + (n0 + mn)] = v;
    }
    __syncthreads();

    // out = h @ W2 + b2 : 16 rows x 2 classes, 32 threads
    if (threadIdx.x < 32) {
        const int row = threadIdx.x >> 1;
        const int c   = threadIdx.x & 1;
        float s = b2[c];
        #pragma unroll 8
        for (int k = 0; k < 128; ++k)
            s += h_lds[row * 128 + k] * W2[k * 2 + c];
        out[(long long)(r0 + row) * 2 + c] = s;
    }
}

// ---------------------------------------------------------------------------
extern "C" void kernel_launch(void* const* d_in, const int* in_sizes, int n_in,
                              void* d_out, int out_size, void* d_ws, size_t ws_size,
                              hipStream_t stream) {
    const int*   x   = (const int*)d_in[0];     // [B, 200] token ids
    const float* emb = (const float*)d_in[1];   // [100000, 128]
    const float* W1  = (const float*)d_in[2];   // [128, 128]
    const float* b1  = (const float*)d_in[3];   // [128]
    const float* W2  = (const float*)d_in[4];   // [128, 2]
    const float* b2  = (const float*)d_in[5];   // [2]
    float* out = (float*)d_out;

    const int L = 200;
    const int B = in_sizes[0] / L;              // 16384

    float* pooled = (float*)d_ws;               // [B, 128] f32 scratch (8 MB)

    gather_pool_kernel<<<B, 128, 0, stream>>>(x, emb, pooled);
    mlp_wmma_kernel<<<B / 16, 256, 0, stream>>>(pooled, W1, b1, W2, b2, out);
}
